// MultiHeadSelfAttention_10299331576560
// MI455X (gfx1250) — matrix operand order repaired and checked by reading, not compiled, not run on hardware
//
#include <hip/hip_runtime.h>
#include <hip/hip_bf16.h>

// ---------------------------------------------------------------------------
// Multi-head self-attention for MI455X (gfx1250, wave32, WMMA 16x16x32 f16)
//   B=4, N=2048, C=1024, H=16, D=64
// Pipeline:
//   1) x: f32 -> f16;  weights: f32 -> f16 TRANSPOSED (N,K) for coalesced
//      B-fragment staging in the GEMM.
//   2) WMMA GEMM x@W+b -> Q,K in (B,H,N,D) f16 ; V in (B,H,D,N) f16 so the
//      attention kernel can stage V^T with vector loads.
//   3) flash-attention, 64-key chunks, double-buffered LDS, online softmax.
//   4) WMMA GEMM ctx@Wo+bo -> d_out f32.
// Workspace: x_h 16MB + 4 weights 8MB + Q/K/V 48MB + ctx 16MB = 88MB.
// ---------------------------------------------------------------------------

typedef _Float16 half_t;
typedef half_t v8h  __attribute__((ext_vector_type(8)));
typedef half_t v16h __attribute__((ext_vector_type(16)));
typedef float  v8f  __attribute__((ext_vector_type(8)));

#define WMMA_F16(a, b, c) \
  __builtin_amdgcn_wmma_f32_16x16x32_f16(false, (a), false, (b), (short)0, (c), false, false)

#define CAT16(lo, hi) __builtin_shufflevector((lo), (hi), 0,1,2,3,4,5,6,7,8,9,10,11,12,13,14,15)

// ---------------------------------------------------------------------------
// f32 -> f16 conversion (grid-stride, vector-friendly)
// ---------------------------------------------------------------------------
__global__ __launch_bounds__(256) void cvt_f32_f16(const float* __restrict__ in,
                                                   half_t* __restrict__ out, size_t n) {
  size_t i = (size_t)blockIdx.x * blockDim.x + threadIdx.x;
  size_t stride = (size_t)gridDim.x * blockDim.x;
  for (; i < n; i += stride) out[i] = (half_t)in[i];
}

// f32 (R,Cc) row-major -> f16 (Cc,R) row-major (transpose via LDS tile)
__global__ __launch_bounds__(256) void cvt_f32_f16_t(const float* __restrict__ in,
                                                     half_t* __restrict__ out,
                                                     int R, int Cc) {
  __shared__ half_t tile[32][33];
  const int c0 = blockIdx.x * 32, r0 = blockIdx.y * 32;
  const int tx = threadIdx.x & 31, ty = threadIdx.x >> 5;  // 32 x 8
#pragma unroll
  for (int i = 0; i < 32; i += 8)
    tile[ty + i][tx] = (half_t)in[(size_t)(r0 + ty + i) * Cc + c0 + tx];
  __syncthreads();
#pragma unroll
  for (int i = 0; i < 32; i += 8)
    out[(size_t)(c0 + ty + i) * R + r0 + tx] = tile[tx][ty + i];
}

// ---------------------------------------------------------------------------
// WMMA GEMM: out[M,Nn] = A[M,K](f16) @ W(K,Nn given TRANSPOSED as Wt[Nn,K]) + bias
// Block tile 128x128, 8 waves (4M x 2N), wave tile 32x64 (2x4 frags).
// Double-buffered LDS. OUT_MODE: 0 = f32 row-major, 1 = f16 (B,H,N,D),
// 2 = f16 (B,H,D,N)  (for V).
// ---------------------------------------------------------------------------
template <int OUT_MODE>
__global__ __launch_bounds__(256) void gemm_f16(const half_t* __restrict__ A,
                                                const half_t* __restrict__ Wt,
                                                const float* __restrict__ bias,
                                                void* __restrict__ out,
                                                int M, int K, int Nn, int Nseq) {
  __shared__ half_t As[2][128][40];  // row stride 80B -> 16B-aligned frags
  __shared__ half_t Ws[2][128][40];  // Ws[n][k] (weights pre-transposed)

  const int tid    = threadIdx.x;
  const int lane   = tid & 31;
  const int wave   = tid >> 5;
  const int laneHi = lane >> 4;
  const int lane16 = lane & 15;
  const int wm     = wave >> 1;  // 0..3
  const int wn     = wave & 1;   // 0..1

  const int row0 = blockIdx.y * 128;
  const int col0 = blockIdx.x * 128;

  const int r   = tid >> 1;          // 0..127 (staging row)
  const int seg = (tid & 1) * 16;    // 0 / 16

  v8f acc[2][4];
#pragma unroll
  for (int mt = 0; mt < 2; ++mt)
#pragma unroll
    for (int nt = 0; nt < 4; ++nt) { v8f z = {}; acc[mt][nt] = z; }

  // prologue: stage k-step 0 into buffer 0 (all 16B vector loads)
  {
    const v8h* sa = (const v8h*)(A + (size_t)(row0 + r) * K + seg);
    *(v8h*)&As[0][r][seg]     = sa[0];
    *(v8h*)&As[0][r][seg + 8] = sa[1];
    const v8h* sw = (const v8h*)(Wt + (size_t)(col0 + r) * K + seg);
    *(v8h*)&Ws[0][r][seg]     = sw[0];
    *(v8h*)&Ws[0][r][seg + 8] = sw[1];
  }
  __syncthreads();

  for (int k0 = 0; k0 < K; k0 += 32) {
    const int buf = (k0 >> 5) & 1;
    // prefetch next k-step into the other buffer (overlaps WMMA burst)
    if (k0 + 32 < K) {
      const v8h* sa = (const v8h*)(A + (size_t)(row0 + r) * K + k0 + 32 + seg);
      *(v8h*)&As[buf ^ 1][r][seg]     = sa[0];
      *(v8h*)&As[buf ^ 1][r][seg + 8] = sa[1];
      const v8h* sw = (const v8h*)(Wt + (size_t)(col0 + r) * K + k0 + 32 + seg);
      *(v8h*)&Ws[buf ^ 1][r][seg]     = sw[0];
      *(v8h*)&Ws[buf ^ 1][r][seg + 8] = sw[1];
    }

    // A fragments (16-bit A 16x32 ISA layout)
    v16h afrag[2];
#pragma unroll
    for (int mt = 0; mt < 2; ++mt) {
      int ar = wm * 32 + mt * 16 + lane16;
      int kb = laneHi ? 8 : 0;
      v8h lo = *(const v8h*)&As[buf][ar][kb];
      v8h hi = *(const v8h*)&As[buf][ar][kb + 16];
      afrag[mt] = CAT16(lo, hi);
    }
    // B fragments: lane = column N; lane<16 K0-7,16-23, lane>=16 K8-15,24-31
#pragma unroll
    for (int nt = 0; nt < 4; ++nt) {
      int n  = wn * 64 + nt * 16 + lane16;
      int kb = laneHi * 8;
      v8h lo = *(const v8h*)&Ws[buf][n][kb];
      v8h hi = *(const v8h*)&Ws[buf][n][kb + 16];
      v16h bfrag = CAT16(lo, hi);
#pragma unroll
      for (int mt = 0; mt < 2; ++mt)
        acc[mt][nt] = WMMA_F16(afrag[mt], bfrag, acc[mt][nt]);
    }
    __syncthreads();
  }

  // Epilogue. C/D layout: VGPR rr, lanes0-15 -> (M=rr,N=lane), lanes16-31 -> (M=rr+8).
#pragma unroll
  for (int mt = 0; mt < 2; ++mt) {
#pragma unroll
    for (int nt = 0; nt < 4; ++nt) {
      int col  = col0 + wn * 64 + nt * 16 + lane16;
      float bv = bias[col];
#pragma unroll
      for (int rr = 0; rr < 8; ++rr) {
        int row   = row0 + wm * 32 + mt * 16 + rr + laneHi * 8;
        float val = acc[mt][nt][rr] + bv;
        if (OUT_MODE == 0) {
          ((float*)out)[(size_t)row * Nn + col] = val;
        } else {
          int b = row / Nseq, n = row - (row / Nseq) * Nseq;
          int h = col >> 6, d = col & 63;
          if (OUT_MODE == 1)  // (B,H,N,D)
            ((half_t*)out)[(((size_t)(b * 16 + h) * Nseq) + n) * 64 + d] = (half_t)val;
          else                 // (B,H,D,N) for V
            ((half_t*)out)[((size_t)(b * 16 + h) * 64 + d) * Nseq + n] = (half_t)val;
        }
      }
    }
  }
}

// ---------------------------------------------------------------------------
// Flash attention: block = one (b,h) x 128 query rows; wave = 16 query rows.
// 64-key chunks, double-buffered K/V staging, online softmax.
// Q in (B,H,N,D); K in (B,H,N,D); V in (B,H,D,N).
// ---------------------------------------------------------------------------
__global__ __launch_bounds__(256) void attention_f16(const half_t* __restrict__ Qg,
                                                     const half_t* __restrict__ Kg,
                                                     const half_t* __restrict__ Vg,
                                                     half_t* __restrict__ ctx_out,
                                                     int Nseq) {
  __shared__ half_t Ks[2][64][72];    // [key][d], row 144B
  __shared__ half_t Vt[2][64][72];    // [d][key], row 144B
  __shared__ half_t Pw[8][16][72];    // per-wave P transpose scratch

  const int tid    = threadIdx.x;
  const int lane   = tid & 31;
  const int wave   = tid >> 5;
  const int laneHi = lane >> 4;
  const int lane16 = lane & 15;

  const int nQChunks = Nseq / 128;             // 16
  const int bh       = blockIdx.x / nQChunks;  // 0..63
  const int qchunk   = blockIdx.x % nQChunks;
  const int qBase    = qchunk * 128 + wave * 16;

  const half_t* Qbh  = Qg + (size_t)bh * Nseq * 64;
  const half_t* Kbh  = Kg + (size_t)bh * Nseq * 64;
  const half_t* VbhT = Vg + (size_t)bh * 64 * Nseq;  // (D, Nseq)

  const int sr   = tid >> 2;         // 0..63 (staging row)
  const int sseg = (tid & 3) * 16;   // 0/16/32/48

  // Q A-fragments, pre-scaled by 1/sqrt(D)=0.125 (exact in f16)
  v16h qfrag[2];
#pragma unroll
  for (int s = 0; s < 2; ++s) {
    int row = qBase + lane16;
    int d0  = s * 32 + (laneHi ? 8 : 0);
    v8h lo  = *(const v8h*)(Qbh + (size_t)row * 64 + d0);
    v8h hi  = *(const v8h*)(Qbh + (size_t)row * 64 + d0 + 16);
    v16h qf = CAT16(lo, hi);
    qfrag[s] = qf * (half_t)0.125f;
  }

  float m_i[8], l_i[8];
  v8f ctx[4];
#pragma unroll
  for (int rr = 0; rr < 8; ++rr) { m_i[rr] = -1e30f; l_i[rr] = 0.0f; }
#pragma unroll
  for (int nt = 0; nt < 4; ++nt) { v8f z = {}; ctx[nt] = z; }

  // prologue: stage chunk 0 (all 16B vector loads)
  {
    const v8h* sk = (const v8h*)(Kbh + (size_t)sr * 64 + sseg);
    *(v8h*)&Ks[0][sr][sseg]     = sk[0];
    *(v8h*)&Ks[0][sr][sseg + 8] = sk[1];
    const v8h* sv = (const v8h*)(VbhT + (size_t)sr * Nseq + sseg);
    *(v8h*)&Vt[0][sr][sseg]     = sv[0];
    *(v8h*)&Vt[0][sr][sseg + 8] = sv[1];
  }
  __syncthreads();

  for (int kc = 0; kc < Nseq; kc += 64) {
    const int buf = (kc >> 6) & 1;
    // prefetch next 64-key chunk
    if (kc + 64 < Nseq) {
      const v8h* sk = (const v8h*)(Kbh + (size_t)(kc + 64 + sr) * 64 + sseg);
      *(v8h*)&Ks[buf ^ 1][sr][sseg]     = sk[0];
      *(v8h*)&Ks[buf ^ 1][sr][sseg + 8] = sk[1];
      const v8h* sv = (const v8h*)(VbhT + (size_t)sr * Nseq + kc + 64 + sseg);
      *(v8h*)&Vt[buf ^ 1][sr][sseg]     = sv[0];
      *(v8h*)&Vt[buf ^ 1][sr][sseg + 8] = sv[1];
    }

    // --- S(16x64) = Qs @ K^T : 4 col tiles x 2 d-steps = 8 WMMAs ---
    v8f sacc[4];
#pragma unroll
    for (int ct = 0; ct < 4; ++ct) {
      v8f z = {};
      sacc[ct] = z;
#pragma unroll
      for (int s = 0; s < 2; ++s) {
        int key = ct * 16 + lane16;
        int d0  = s * 32 + laneHi * 8;
        v8h lo  = *(const v8h*)&Ks[buf][key][d0];
        v8h hi  = *(const v8h*)&Ks[buf][key][d0 + 16];
        v16h bfrag = CAT16(lo, hi);
        sacc[ct] = WMMA_F16(qfrag[s], bfrag, sacc[ct]);
      }
    }

    // --- online softmax over 64 new columns (one reduction pair per row) ---
    float alpha[8];
#pragma unroll
    for (int rr = 0; rr < 8; ++rr) {
      float mx = fmaxf(fmaxf(sacc[0][rr], sacc[1][rr]),
                       fmaxf(sacc[2][rr], sacc[3][rr]));
#pragma unroll
      for (int off = 8; off > 0; off >>= 1)
        mx = fmaxf(mx, __shfl_xor(mx, off, 16));
      float mnew = fmaxf(m_i[rr], mx);
      float a    = __expf(m_i[rr] - mnew);
      float p0 = __expf(sacc[0][rr] - mnew);
      float p1 = __expf(sacc[1][rr] - mnew);
      float p2 = __expf(sacc[2][rr] - mnew);
      float p3 = __expf(sacc[3][rr] - mnew);
      float rs = (p0 + p1) + (p2 + p3);
#pragma unroll
      for (int off = 8; off > 0; off >>= 1)
        rs += __shfl_xor(rs, off, 16);
      m_i[rr]   = mnew;
      l_i[rr]   = l_i[rr] * a + rs;
      alpha[rr] = a;
      sacc[0][rr] = p0; sacc[1][rr] = p1; sacc[2][rr] = p2; sacc[3][rr] = p3;
    }
#pragma unroll
    for (int nt = 0; nt < 4; ++nt)
#pragma unroll
      for (int rr = 0; rr < 8; ++rr) ctx[nt][rr] *= alpha[rr];

    // --- P (C-layout) -> per-wave LDS -> A-fragment layout ---
#pragma unroll
    for (int ct = 0; ct < 4; ++ct)
#pragma unroll
      for (int rr = 0; rr < 8; ++rr)
        Pw[wave][rr + 8 * laneHi][ct * 16 + lane16] = (half_t)sacc[ct][rr];

    v16h pfrag[2];
#pragma unroll
    for (int s = 0; s < 2; ++s) {
      int kb = s * 32 + (laneHi ? 8 : 0);
      v8h lo = *(const v8h*)&Pw[wave][lane16][kb];
      v8h hi = *(const v8h*)&Pw[wave][lane16][kb + 16];
      pfrag[s] = CAT16(lo, hi);
    }

    // --- ctx(16x64) += P(16x64) @ V(64x64) : 4 n-tiles x 2 k-steps ---
#pragma unroll
    for (int nt = 0; nt < 4; ++nt) {
      int d = nt * 16 + lane16;
#pragma unroll
      for (int s = 0; s < 2; ++s) {
        int kb = s * 32 + laneHi * 8;
        v8h blo = *(const v8h*)&Vt[buf][d][kb];
        v8h bhi = *(const v8h*)&Vt[buf][d][kb + 16];
        v16h vfrag = CAT16(blo, bhi);
        ctx[nt] = WMMA_F16(pfrag[s], vfrag, ctx[nt]);
      }
    }
    __syncthreads();
  }

  // --- finalize: ctx / l, write f16 (B,N,C) merged-head layout ---
  const int b = bh >> 4, h = bh & 15;
#pragma unroll
  for (int nt = 0; nt < 4; ++nt) {
#pragma unroll
    for (int rr = 0; rr < 8; ++rr) {
      int row = rr + 8 * laneHi;
      int n   = qBase + row;
      int c   = h * 64 + nt * 16 + lane16;
      ctx_out[((size_t)(b * Nseq + n)) * 1024 + c] = (half_t)(ctx[nt][rr] / l_i[rr]);
    }
  }
}

// ---------------------------------------------------------------------------
// Host launcher
// ---------------------------------------------------------------------------
extern "C" void kernel_launch(void* const* d_in, const int* in_sizes, int n_in,
                              void* d_out, int out_size, void* d_ws, size_t ws_size,
                              hipStream_t stream) {
  const float* x  = (const float*)d_in[0];
  const float* wq = (const float*)d_in[1];
  const float* bq = (const float*)d_in[2];
  const float* wk = (const float*)d_in[3];
  const float* bk = (const float*)d_in[4];
  const float* wv = (const float*)d_in[5];
  const float* bv = (const float*)d_in[6];
  const float* wo = (const float*)d_in[7];
  const float* bo = (const float*)d_in[8];

  const int Bc = 4, Nseq = 2048, C = 1024;
  const int M = Bc * Nseq;  // 8192

  char* p = (char*)d_ws;
  half_t* xh   = (half_t*)p; p += (size_t)M * C * sizeof(half_t);
  half_t* wqt  = (half_t*)p; p += (size_t)C * C * sizeof(half_t);
  half_t* wkt  = (half_t*)p; p += (size_t)C * C * sizeof(half_t);
  half_t* wvt  = (half_t*)p; p += (size_t)C * C * sizeof(half_t);
  half_t* wot  = (half_t*)p; p += (size_t)C * C * sizeof(half_t);
  half_t* qh   = (half_t*)p; p += (size_t)M * C * sizeof(half_t);
  half_t* kh   = (half_t*)p; p += (size_t)M * C * sizeof(half_t);
  half_t* vh   = (half_t*)p; p += (size_t)M * C * sizeof(half_t);   // (B,H,D,N)
  half_t* ctxh = (half_t*)p; p += (size_t)M * C * sizeof(half_t);

  // 1) conversions: x plain, weights transposed to (N,K)
  cvt_f32_f16<<<4096, 256, 0, stream>>>(x, xh, (size_t)M * C);
  dim3 tg(C / 32, C / 32);  // (32, 32)
  cvt_f32_f16_t<<<tg, 256, 0, stream>>>(wq, wqt, C, C);
  cvt_f32_f16_t<<<tg, 256, 0, stream>>>(wk, wkt, C, C);
  cvt_f32_f16_t<<<tg, 256, 0, stream>>>(wv, wvt, C, C);
  cvt_f32_f16_t<<<tg, 256, 0, stream>>>(wo, wot, C, C);

  // 2) QKV projections
  dim3 gg(C / 128, M / 128);  // (8, 64)
  gemm_f16<1><<<gg, 256, 0, stream>>>(xh, wqt, bq, (void*)qh, M, C, C, Nseq);
  gemm_f16<1><<<gg, 256, 0, stream>>>(xh, wkt, bk, (void*)kh, M, C, C, Nseq);
  gemm_f16<2><<<gg, 256, 0, stream>>>(xh, wvt, bv, (void*)vh, M, C, C, Nseq);

  // 3) flash attention -> ctx (B,N,C) f16
  attention_f16<<<Bc * 16 * (Nseq / 128), 256, 0, stream>>>(qh, kh, vh, ctxh, Nseq);

  // 4) output projection -> f32 d_out
  gemm_f16<0><<<gg, 256, 0, stream>>>(ctxh, wot, bo, d_out, M, C, C, Nseq);
}
